// DeltaNet_31877247271544
// MI455X (gfx1250) — compile-verified, matrix operand order checked by reading, not run on hardware
//
#include <hip/hip_runtime.h>
#include <hip/hip_bf16.h>
#include <math.h>

// ---------------------------------------------------------------------------
// Types / WMMA helpers (CDNA5 gfx1250, wave32)
// ---------------------------------------------------------------------------
typedef __attribute__((ext_vector_type(16))) _Float16 v16h;
typedef __attribute__((ext_vector_type(8)))  _Float16 v8h;
typedef __attribute__((ext_vector_type(8)))  float    v8f;

#define WMMA_F16(a, b, c) \
  __builtin_amdgcn_wmma_f32_16x16x32_f16(false, (a), false, (b), (short)0, (c), false, false)

// Load a 16x32 (rows x K-halves) fragment from an LDS row-major array.
// Per ISA 7.12.2: lane L<16 holds row M=L with K in {sub..sub+7} U {16+sub..},
// sub = (L>=16)?8:0.  Two 16B LDS reads per lane.
__device__ __forceinline__ v16h ldfrag(const _Float16* base, int stride, int row0, int k0) {
  int lane = threadIdx.x & 31;
  int r    = row0 + (lane & 15);
  int sub  = (lane >> 4) << 3;
  const _Float16* p = base + r * stride + k0 + sub;
  v8h lo = *(const v8h*)p;
  v8h hi = *(const v8h*)(p + 16);
  v16h f;
#pragma unroll
  for (int i = 0; i < 8; ++i) { f[i] = lo[i]; f[8 + i] = hi[i]; }
  return f;
}

__device__ __forceinline__ float wave_sum(float s) {
#pragma unroll
  for (int off = 16; off > 0; off >>= 1) s += __shfl_xor(s, off);
  return s;
}

__device__ __forceinline__ float sigmoidf_(float x) { return 1.0f / (1.0f + expf(-x)); }

// ---------------------------------------------------------------------------
// Problem constants
// ---------------------------------------------------------------------------
#define BB   4
#define LL   2048
#define HH   1024
#define NHH  8
#define DKK  128
#define TOK  (BB * LL)        // 8192
#define CMAXF 0.2f

// ---------------------------------------------------------------------------
// 1) f32 -> f16 convert (row-major copy)
// ---------------------------------------------------------------------------
__global__ __launch_bounds__(256) void k_convert_f16(const float* __restrict__ x,
                                                     _Float16* __restrict__ y, int n) {
  int i = blockIdx.x * 256 + threadIdx.x;
  if (i < n) y[i] = (_Float16)x[i];
}

// 2) transpose + convert: W[K][N] f32 -> WT[N][K] f16   (grid: (N/32, K/32), block (32,8))
__global__ void k_transpose_f16(const float* __restrict__ W, _Float16* __restrict__ WT,
                                int K, int N) {
  __shared__ float tile[32][33];
  int kb = blockIdx.y * 32, nb = blockIdx.x * 32;
  int tx = threadIdx.x, ty = threadIdx.y;
  for (int i = ty; i < 32; i += 8) tile[i][tx] = W[(long)(kb + i) * N + nb + tx];
  __syncthreads();
  for (int i = ty; i < 32; i += 8) WT[(long)(nb + i) * K + kb + tx] = (_Float16)tile[tx][i];
}

// ---------------------------------------------------------------------------
// 3) WMMA GEMM:  C[M][N] (f32) = A16[M][K] * (Bt16[N][K])^T, tile 128x128, 8 waves
//    mode 0: plain store; mode 1: +bias[n], exact GELU
// ---------------------------------------------------------------------------
__global__ __launch_bounds__(256) void k_wmma_gemm(const _Float16* __restrict__ A,
                                                   const _Float16* __restrict__ Bt,
                                                   float* __restrict__ C,
                                                   const float* __restrict__ bias,
                                                   int M, int N, int K, int mode) {
  __shared__ _Float16 As[128 * 32];
  __shared__ _Float16 Bs[128 * 32];
  int tid = threadIdx.x, wave = tid >> 5, lane = tid & 31;
  int bm = blockIdx.y * 128, bn = blockIdx.x * 128;
  int wm = (wave >> 2) * 64;       // 0 / 64
  int wn = (wave & 3) * 32;        // 0/32/64/96
  int cAdd = (lane >> 4) * 8;

  v8f acc[4][2];
#pragma unroll
  for (int a = 0; a < 4; ++a)
#pragma unroll
    for (int b = 0; b < 2; ++b) acc[a][b] = (v8f){};

  int lr = tid >> 1;               // 0..127
  int lc = (tid & 1) * 16;         // 0 / 16

  for (int k0 = 0; k0 < K; k0 += 32) {
    const _Float16* ag = A + (long)(bm + lr) * K + k0 + lc;
    const _Float16* bg = Bt + (long)(bn + lr) * K + k0 + lc;
    *(v8h*)(As + lr * 32 + lc)     = *(const v8h*)ag;
    *(v8h*)(As + lr * 32 + lc + 8) = *(const v8h*)(ag + 8);
    *(v8h*)(Bs + lr * 32 + lc)     = *(const v8h*)bg;
    *(v8h*)(Bs + lr * 32 + lc + 8) = *(const v8h*)(bg + 8);
    __syncthreads();

    v16h bf0 = ldfrag(Bs, 32, wn, 0);
    v16h bf1 = ldfrag(Bs, 32, wn + 16, 0);
#pragma unroll
    for (int mi = 0; mi < 4; ++mi) {
      v16h af = ldfrag(As, 32, wm + 16 * mi, 0);
      acc[mi][0] = WMMA_F16(af, bf0, acc[mi][0]);
      acc[mi][1] = WMMA_F16(af, bf1, acc[mi][1]);
    }
    __syncthreads();
  }

#pragma unroll
  for (int mi = 0; mi < 4; ++mi)
#pragma unroll
    for (int ni = 0; ni < 2; ++ni) {
      int r0 = bm + wm + 16 * mi + cAdd;
      int c0 = bn + wn + 16 * ni + (lane & 15);
#pragma unroll
      for (int i = 0; i < 8; ++i) {
        float v = acc[mi][ni][i];
        if (mode == 1) {
          v += bias[c0];
          v = 0.5f * v * (1.0f + erff(v * 0.70710678118f));
        }
        C[(long)(r0 + i) * N + c0] = v;
      }
    }
}

// ---------------------------------------------------------------------------
// 4) causal conv (K=4) + SiLU; in [B][L][H] f32 -> out [B*NH][L][128] f32
// ---------------------------------------------------------------------------
__global__ __launch_bounds__(256) void k_conv_silu(const float* __restrict__ x,
                                                   const float* __restrict__ w,
                                                   float* __restrict__ out) {
  int idx = blockIdx.x * 256 + threadIdx.x;       // B*L*H
  int c = idx & (HH - 1);
  int t = idx >> 10;
  int l = t & (LL - 1);
  int b = t >> 11;
  float acc = 0.f;
#pragma unroll
  for (int i = 0; i < 4; ++i) {
    int ls = l + i - 3;
    if (ls >= 0) acc += x[(long)(b * LL + ls) * HH + c] * w[c * 4 + i];
  }
  float s = acc * sigmoidf_(acc);
  int h = c >> 7, d = c & 127;
  out[((long)(b * NHH + h) * LL + l) * DKK + d] = s;
}

// 5) per-row L2 norm over 128 (in place).  One wave per row.
__global__ __launch_bounds__(256) void k_l2norm(float* __restrict__ x) {
  int row = blockIdx.x * 8 + (threadIdx.x >> 5);
  int lane = threadIdx.x & 31;
  float* p = x + (long)row * 128 + lane * 4;
  float4 v = *(float4*)p;
  float s = v.x * v.x + v.y * v.y + v.z * v.z + v.w * v.w;
  s = wave_sum(s);
  float r = rsqrtf(s + 1e-6f);
  v.x *= r; v.y *= r; v.z *= r; v.w *= r;
  *(float4*)p = v;
}

// ---------------------------------------------------------------------------
// 6) gating: beta = sig(hs@Wb); floor = sig(hs@floorW+b)*CMAX;
//    fus = g@W2 + b2 -> p, ctx softmax; one block per token
// ---------------------------------------------------------------------------
__global__ __launch_bounds__(256) void k_gates(const float* __restrict__ hs,
                                               const float* __restrict__ g,
                                               const float* __restrict__ Wb,
                                               const float* __restrict__ floorW,
                                               const float* __restrict__ floorb,
                                               const float* __restrict__ W2,
                                               const float* __restrict__ b2,
                                               float* __restrict__ betaBuf,
                                               float* __restrict__ coefBuf) {
  int tok = blockIdx.x;
  int wave = threadIdx.x >> 5, lane = threadIdx.x & 31;
  __shared__ float vals[48];
  const float* hrow = hs + (long)tok * HH;
  const float* grow = g + (long)tok * 2048;

  for (int d = wave; d < 48; d += 8) {
    float s = 0.f;
    if (d < 8) {
      for (int k = lane; k < HH; k += 32) s += hrow[k] * Wb[k * 8 + d];
    } else if (d < 16) {
      int h = d - 8;
      for (int k = lane; k < HH; k += 32) s += hrow[k] * floorW[k * 8 + h];
    } else {
      int j = d - 16;
      for (int k = lane; k < 2048; k += 32) s += grow[k] * W2[k * 32 + j];
    }
    s = wave_sum(s);
    if (lane == 0) {
      if (d < 8)       betaBuf[(long)tok * 8 + d] = sigmoidf_(s);
      else if (d < 16) vals[d] = sigmoidf_(s + floorb[d - 8]) * CMAXF;
      else             vals[d] = s + b2[d - 16];
    }
  }
  __syncthreads();
  if (threadIdx.x < 8) {
    int h = threadIdx.x;
    float fl = vals[8 + h];
    float f0 = vals[16 + h * 4 + 0], f1 = vals[16 + h * 4 + 1];
    float f2 = vals[16 + h * 4 + 2], f3 = vals[16 + h * 4 + 3];
    float p = sigmoidf_(f3) * (1.0f - fl);
    float m = fmaxf(f0, fmaxf(f1, f2));
    float e0 = expf(f0 - m), e1 = expf(f1 - m), e2 = expf(f2 - m);
    float sc = (1.0f - p) / (e0 + e1 + e2);
    float* cf = coefBuf + (long)tok * 32 + h * 4;
    cf[0] = e0 * sc; cf[1] = e1 * sc; cf[2] = e2 * sc; cf[3] = p;
  }
}

// ---------------------------------------------------------------------------
// 7) delta-rule chunkwise scan.  One block per (b,h), 8 waves, 64 chunks of 32.
//    S (128x128 f32) + ST (f16 transpose) live in dynamic LDS.
// ---------------------------------------------------------------------------
#define DELTA_SMEM 174080

__global__ __launch_bounds__(256) void k_delta(const float* __restrict__ qn,
                                               const float* __restrict__ kn,
                                               const float* __restrict__ vh,
                                               const float* __restrict__ betaBuf,
                                               float* __restrict__ dout) {
  extern __shared__ char smem[];
  float*    S    = (float*)(smem);              // 128x128 f32
  _Float16* ST   = (_Float16*)(smem + 65536);   // 128x128 f16 (ST[n][k]=S[k][n])
  _Float16* q16  = (_Float16*)(smem + 98304);   // 32x128
  _Float16* k16  = (_Float16*)(smem + 106496);  // 32x128
  _Float16* kT   = (_Float16*)(smem + 114688);  // 128x32
  _Float16* vT   = (_Float16*)(smem + 122880);  // 128x32  (v*beta)^T
  _Float16* uT   = (_Float16*)(smem + 131072);  // 128x32  u_new^T
  _Float16* w16  = (_Float16*)(smem + 139264);  // 32x128
  float*    uf   = (float*)(smem + 147456);     // 32x128
  float*    attn = (float*)(smem + 163840);     // 32x32
  _Float16* a16  = (_Float16*)(smem + 167936);  // 32x32
  _Float16* ab16 = (_Float16*)(smem + 169984);  // 32x32 (attn * beta[col])
  _Float16* al16 = (_Float16*)(smem + 172032);  // 32x32 attn_local

  int bh = blockIdx.x;
  int b = bh >> 3, h = bh & 7;
  int tid = threadIdx.x, wave = tid >> 5, lane = tid & 31;
  int cAdd = (lane >> 4) * 8;
  const long base = (long)bh * LL * DKK;
  const float* bBase = betaBuf + (long)b * LL * 8 + h;

  for (int i = tid; i < 128 * 128; i += 256) S[i] = 0.f;
  for (int i = tid; i < 128 * 128; i += 256) ST[i] = (_Float16)0.f;
  __syncthreads();

  for (int c = 0; c < 64; ++c) {
    int l0 = c * 32;
    // ---- stage chunk: q16/k16/kT/vT(beta-scaled) ----
    {
      int e0 = tid * 16;
      int row = e0 >> 7, col = e0 & 127;
      float beta = bBase[(long)(l0 + row) * 8];
      const float* qp = qn + base + (long)(l0 + row) * 128 + col;
      const float* kp = kn + base + (long)(l0 + row) * 128 + col;
      const float* vp = vh + base + (long)(l0 + row) * 128 + col;
#pragma unroll
      for (int j = 0; j < 16; ++j) {
        float kv = kp[j];
        q16[row * 128 + col + j] = (_Float16)qp[j];
        k16[row * 128 + col + j] = (_Float16)kv;
        kT[(col + j) * 32 + row] = (_Float16)kv;
        vT[(col + j) * 32 + row] = (_Float16)(vp[j] * beta);
      }
    }
    __syncthreads();

    // ---- attn_pre = -(beta*k) k^T (waves 0-3); attn_local = q k^T (waves 4-7)
    if (wave < 4) {
      int m0 = (wave >> 1) * 16, n0 = (wave & 1) * 16;
      _Float16 bs = (_Float16)bBase[(long)(l0 + m0 + (lane & 15)) * 8];
      v8f acc = (v8f){};
#pragma unroll
      for (int ks = 0; ks < 4; ++ks) {
        v16h a = ldfrag(k16, 128, m0, ks * 32);
#pragma unroll
        for (int j = 0; j < 16; ++j) a[j] = a[j] * bs;
        v16h bb = ldfrag(k16, 128, n0, ks * 32);
        acc = WMMA_F16(a, bb, acc);
      }
#pragma unroll
      for (int i = 0; i < 8; ++i) {
        int M = m0 + cAdd + i, N = n0 + (lane & 15);
        attn[M * 32 + N] = (N >= M) ? 0.f : -acc[i];
      }
    } else {
      int w2 = wave - 4;
      int m0 = (w2 >> 1) * 16, n0 = (w2 & 1) * 16;
      v8f acc = (v8f){};
#pragma unroll
      for (int ks = 0; ks < 4; ++ks)
        acc = WMMA_F16(ldfrag(q16, 128, m0, ks * 32), ldfrag(k16, 128, n0, ks * 32), acc);
#pragma unroll
      for (int i = 0; i < 8; ++i) {
        int M = m0 + cAdd + i, N = n0 + (lane & 15);
        al16[M * 32 + N] = (N > M) ? (_Float16)0.f : (_Float16)acc[i];
      }
    }
    __syncthreads();

    // ---- forward substitution (fp32, sequential in i) ----
    for (int i = 1; i < 32; ++i) {
      float s = 0.f;
      if (tid < i) {
        for (int j = tid + 1; j < i; ++j) s += attn[i * 32 + j] * attn[j * 32 + tid];
      }
      __syncthreads();
      if (tid < i) attn[i * 32 + tid] += s;
      __syncthreads();
    }
    if (tid < 32) attn[tid * 32 + tid] += 1.0f;
    __syncthreads();
    for (int e = tid; e < 1024; e += 256) {
      int t = e & 31;
      float f = attn[e];
      a16[e]  = (_Float16)f;
      ab16[e] = (_Float16)(f * bBase[(long)(l0 + t) * 8]);
    }
    __syncthreads();

    // ---- u = attn @ (v*beta);  w = (attn*beta) @ k  (K=32, one WMMA) ----
#pragma unroll
    for (int tt = 0; tt < 2; ++tt) {
      int tile = wave * 2 + tt;
      int m0 = (tile >> 3) * 16, n0 = (tile & 7) * 16;
      v8f acc = (v8f){};
      acc = WMMA_F16(ldfrag(a16, 32, m0, 0), ldfrag(vT, 32, n0, 0), acc);
#pragma unroll
      for (int i = 0; i < 8; ++i) uf[(m0 + cAdd + i) * 128 + n0 + (lane & 15)] = acc[i];
      v8f ac2 = (v8f){};
      ac2 = WMMA_F16(ldfrag(ab16, 32, m0, 0), ldfrag(kT, 32, n0, 0), ac2);
#pragma unroll
      for (int i = 0; i < 8; ++i)
        w16[(m0 + cAdd + i) * 128 + n0 + (lane & 15)] = (_Float16)ac2[i];
    }
    __syncthreads();

    // ---- u_new = u - w @ S ; store uT ----
#pragma unroll
    for (int tt = 0; tt < 2; ++tt) {
      int tile = wave * 2 + tt;
      int m0 = (tile >> 3) * 16, n0 = (tile & 7) * 16;
      v8f acc = (v8f){};
#pragma unroll
      for (int ks = 0; ks < 4; ++ks)
        acc = WMMA_F16(ldfrag(w16, 128, m0, ks * 32), ldfrag(ST, 128, n0, ks * 32), acc);
#pragma unroll
      for (int i = 0; i < 8; ++i) {
        int M = m0 + cAdd + i, N = n0 + (lane & 15);
        uT[N * 32 + M] = (_Float16)(uf[M * 128 + N] - acc[i]);
      }
    }
    __syncthreads();

    // ---- o = q @ S + attn_local @ u_new ; write to global ----
#pragma unroll
    for (int tt = 0; tt < 2; ++tt) {
      int tile = wave * 2 + tt;
      int m0 = (tile >> 3) * 16, n0 = (tile & 7) * 16;
      v8f acc = (v8f){};
#pragma unroll
      for (int ks = 0; ks < 4; ++ks)
        acc = WMMA_F16(ldfrag(q16, 128, m0, ks * 32), ldfrag(ST, 128, n0, ks * 32), acc);
      acc = WMMA_F16(ldfrag(al16, 32, m0, 0), ldfrag(uT, 32, n0, 0), acc);
#pragma unroll
      for (int i = 0; i < 8; ++i)
        dout[base + (long)(l0 + m0 + cAdd + i) * 128 + n0 + (lane & 15)] = acc[i];
    }
    __syncthreads();

    // ---- S += k^T @ u_new ; refresh ST ----
#pragma unroll
    for (int tt = 0; tt < 8; ++tt) {
      int tile = wave * 8 + tt;
      int m0 = (tile >> 3) * 16, n0 = (tile & 7) * 16;
      v8f acc;
#pragma unroll
      for (int i = 0; i < 8; ++i) acc[i] = S[(m0 + cAdd + i) * 128 + n0 + (lane & 15)];
      acc = WMMA_F16(ldfrag(kT, 32, m0, 0), ldfrag(uT, 32, n0, 0), acc);
#pragma unroll
      for (int i = 0; i < 8; ++i) {
        int M = m0 + cAdd + i, N = n0 + (lane & 15);
        S[M * 128 + N]  = acc[i];
        ST[N * 128 + M] = (_Float16)acc[i];
      }
    }
    __syncthreads();
  }
}

// ---------------------------------------------------------------------------
// 8) FIR(3)/FIR(31) + gated mix -> o[B*NH][L][128]
// ---------------------------------------------------------------------------
__global__ __launch_bounds__(256) void k_mix(const float* __restrict__ vh,
                                             const float* __restrict__ del,
                                             const float* __restrict__ coef,
                                             const float* __restrict__ firS,
                                             const float* __restrict__ firL,
                                             float* __restrict__ o) {
  int idx = blockIdx.x * 256 + threadIdx.x;   // 32*2048*128
  int d  = idx & 127;
  int l  = (idx >> 7) & (LL - 1);
  int bh = idx >> 18;
  int b = bh >> 3, h = bh & 7;
  const float* vcol = vh + (long)bh * LL * 128 + d;
  float ss = 0.f;
#pragma unroll
  for (int i = 0; i < 3; ++i) {
    int ls = l + i - 2;
    if (ls >= 0) ss += vcol[(long)ls * 128] * firS[(h * 128 + d) * 3 + i];
  }
  float sl = 0.f;
#pragma unroll
  for (int i = 0; i < 31; ++i) {
    int ls = l + i - 30;
    if (ls >= 0) sl += vcol[(long)ls * 128] * firL[(h * 128 + d) * 31 + i];
  }
  const float* cf = coef + ((long)(b * LL + l)) * 32 + h * 4;
  o[idx] = cf[0] * ss + cf[1] * sl + cf[2] * del[idx] + cf[3] * vh[idx];
}

// ---------------------------------------------------------------------------
// 9) double RMS over 128 + convert to f16 token-major [tok][1024]
// ---------------------------------------------------------------------------
__global__ __launch_bounds__(256) void k_rms2(const float* __restrict__ o,
                                              const float* __restrict__ w1,
                                              const float* __restrict__ w2,
                                              _Float16* __restrict__ o16) {
  int row = blockIdx.x * 8 + (threadIdx.x >> 5);   // over [32][2048]
  int lane = threadIdx.x & 31;
  const float* p = o + (long)row * 128 + lane * 4;
  float4 v = *(const float4*)p;
  float s = v.x * v.x + v.y * v.y + v.z * v.z + v.w * v.w;
  s = wave_sum(s);
  float r1 = rsqrtf(s * (1.0f / 128.0f) + 1e-5f);
  float y[4];
  int d0 = lane * 4;
  y[0] = v.x * r1 * w1[d0 + 0]; y[1] = v.y * r1 * w1[d0 + 1];
  y[2] = v.z * r1 * w1[d0 + 2]; y[3] = v.w * r1 * w1[d0 + 3];
  float s2 = y[0] * y[0] + y[1] * y[1] + y[2] * y[2] + y[3] * y[3];
  s2 = wave_sum(s2);
  float r2 = rsqrtf(s2 * (1.0f / 128.0f) + 1e-5f);
  int bh = row >> 11, l = row & (LL - 1);
  int b = bh >> 3, h = bh & 7;
  long out = (long)(b * LL + l) * HH + h * 128 + d0;
#pragma unroll
  for (int j = 0; j < 4; ++j) o16[out + j] = (_Float16)(y[j] * r2 * w2[d0 + j]);
}

// ---------------------------------------------------------------------------
// Host-side launch
// ---------------------------------------------------------------------------
extern "C" void kernel_launch(void* const* d_in, const int* in_sizes, int n_in,
                              void* d_out, int out_size, void* d_ws, size_t ws_size,
                              hipStream_t stream) {
  (void)in_sizes; (void)n_in; (void)out_size; (void)ws_size;
  const float* hs      = (const float*)d_in[0];
  const float* Wq      = (const float*)d_in[1];
  const float* Wk      = (const float*)d_in[2];
  const float* Wv      = (const float*)d_in[3];
  const float* Wb      = (const float*)d_in[4];
  const float* convq   = (const float*)d_in[5];
  const float* convk   = (const float*)d_in[6];
  const float* convv   = (const float*)d_in[7];
  const float* firS    = (const float*)d_in[8];
  const float* firL    = (const float*)d_in[9];
  const float* floorW  = (const float*)d_in[10];
  const float* floorb  = (const float*)d_in[11];
  const float* fusW1   = (const float*)d_in[12];
  const float* fusb1   = (const float*)d_in[13];
  const float* fusW2   = (const float*)d_in[14];
  const float* fusb2   = (const float*)d_in[15];
  const float* normPF  = (const float*)d_in[16];
  const float* normO   = (const float*)d_in[17];
  const float* Wo      = (const float*)d_in[18];
  float* out = (float*)d_out;

  // ---- workspace carve-up (bytes) ----
  char* ws = (char*)d_ws;
  size_t off = 0;
  _Float16* hs16 = (_Float16*)(ws + off); off += (size_t)TOK * HH * 2;        // 16 MB
  _Float16* WqT  = (_Float16*)(ws + off); off += (size_t)HH * HH * 2;         // 2 MB
  _Float16* WkT  = (_Float16*)(ws + off); off += (size_t)HH * HH * 2;
  _Float16* WvT  = (_Float16*)(ws + off); off += (size_t)HH * HH * 2;
  _Float16* W1T  = (_Float16*)(ws + off); off += (size_t)2048 * HH * 2;       // 4 MB
  _Float16* WoT  = (_Float16*)(ws + off); off += (size_t)HH * HH * 2;
  float* q_raw   = (float*)(ws + off);    off += (size_t)TOK * HH * 4;        // 32 MB
  float* k_raw   = (float*)(ws + off);    off += (size_t)TOK * HH * 4;
  float* v_raw   = (float*)(ws + off);    off += (size_t)TOK * HH * 4;
  float* gelu    = (float*)(ws + off);    off += (size_t)TOK * 2048 * 4;      // 64 MB
  float* qn      = (float*)(ws + off);    off += (size_t)TOK * HH * 4;
  float* kn      = (float*)(ws + off);    off += (size_t)TOK * HH * 4;
  float* vh      = (float*)(ws + off);    off += (size_t)TOK * HH * 4;
  float* betaBuf = (float*)(ws + off);    off += (size_t)TOK * 8 * 4;
  float* coefBuf = (float*)(ws + off);    off += (size_t)TOK * 32 * 4;
  // aliased buffers (lifetimes don't overlap):
  float* deltaB  = gelu;                    // 32 MB, used after gating done
  float* o_mix   = gelu + (size_t)TOK * HH; // next 32 MB of gelu region
  _Float16* o16  = (_Float16*)q_raw;        // raw q unused after conv

  dim3 tb(32, 8);
  // 1) converts / transposes
  k_convert_f16<<<(TOK * HH) / 256, 256, 0, stream>>>(hs, hs16, TOK * HH);
  k_transpose_f16<<<dim3(HH / 32, HH / 32), tb, 0, stream>>>(Wq, WqT, HH, HH);
  k_transpose_f16<<<dim3(HH / 32, HH / 32), tb, 0, stream>>>(Wk, WkT, HH, HH);
  k_transpose_f16<<<dim3(HH / 32, HH / 32), tb, 0, stream>>>(Wv, WvT, HH, HH);
  k_transpose_f16<<<dim3(2048 / 32, HH / 32), tb, 0, stream>>>(fusW1, W1T, HH, 2048);
  k_transpose_f16<<<dim3(HH / 32, HH / 32), tb, 0, stream>>>(Wo, WoT, HH, HH);

  // 2) projections (WMMA GEMMs)
  dim3 gQKV(HH / 128, TOK / 128);
  k_wmma_gemm<<<gQKV, 256, 0, stream>>>(hs16, WqT, q_raw, nullptr, TOK, HH, HH, 0);
  k_wmma_gemm<<<gQKV, 256, 0, stream>>>(hs16, WkT, k_raw, nullptr, TOK, HH, HH, 0);
  k_wmma_gemm<<<gQKV, 256, 0, stream>>>(hs16, WvT, v_raw, nullptr, TOK, HH, HH, 0);
  k_wmma_gemm<<<dim3(2048 / 128, TOK / 128), 256, 0, stream>>>(hs16, W1T, gelu, fusb1,
                                                               TOK, 2048, HH, 1);

  // 3) gating coefficients (needs gelu buffer before it is reused)
  k_gates<<<TOK, 256, 0, stream>>>(hs, gelu, Wb, floorW, floorb, fusW2, fusb2,
                                   betaBuf, coefBuf);

  // 4) causal conv + SiLU -> [bh][L][128]
  k_conv_silu<<<(TOK * HH) / 256, 256, 0, stream>>>(q_raw, convq, qn);
  k_conv_silu<<<(TOK * HH) / 256, 256, 0, stream>>>(k_raw, convk, kn);
  k_conv_silu<<<(TOK * HH) / 256, 256, 0, stream>>>(v_raw, convv, vh);

  // 5) L2-normalize q,k rows
  k_l2norm<<<(32 * LL) / 8, 256, 0, stream>>>(qn);
  k_l2norm<<<(32 * LL) / 8, 256, 0, stream>>>(kn);

  // 6) delta rule (WMMA, S in LDS) -> deltaB (aliases gelu)
  k_delta<<<32, 256, DELTA_SMEM, stream>>>(qn, kn, vh, betaBuf, deltaB);

  // 7) FIR + gated mix
  k_mix<<<(32 * LL * 128) / 256, 256, 0, stream>>>(vh, deltaB, coefBuf, firS, firL, o_mix);

  // 8) double RMS + f16 convert (token-major)
  k_rms2<<<(32 * LL) / 8, 256, 0, stream>>>(o_mix, normPF, normO, o16);

  // 9) output projection -> d_out
  k_wmma_gemm<<<dim3(HH / 128, TOK / 128), 256, 0, stream>>>(o16, WoT, out, nullptr,
                                                             TOK, HH, HH, 0);
}
